// IntrospectMambaWrapper_87376814670397
// MI455X (gfx1250) — matrix operand order repaired and checked by reading, not compiled, not run on hardware
//
#include <hip/hip_runtime.h>
#include <hip/hip_bf16.h>

// ---------------------------------------------------------------------------
// Mamba forward for MI455X (gfx1250, wave32).
//  - All GEMMs hoisted out of the scan and run as batched WMMA bf16 GEMMs
//    over the 4096 tokens (v_wmma_f32_16x16x32_bf16, f32 accumulation).
//  - Scan kernel: thread per (b,d,s) = 131072 threads (4096 waves), one f32
//    state per lane, y-readout reduced across the 16 state lanes via
//    __shfl_xor; next-step operands prefetched (global_prefetch_b8).
//    Trajectory stores are 128B-contiguous per wave (537MB -> HBM bound).
// ---------------------------------------------------------------------------

typedef __attribute__((ext_vector_type(8)))  __bf16 v8bf;
typedef __attribute__((ext_vector_type(16))) __bf16 v16bf;
typedef __attribute__((ext_vector_type(8)))  float  v8f;

#define NTOK   4096   // B*L tokens
#define DMODEL 1024
#define DINNER 2048
#define DSTATE 16
#define DCONV  4
#define DTRANK 64
#define SEQB   4
#define SEQL   1024
#define XDBW   96     // DT_RANK + 2*D_STATE

union AFrag { v16bf v; v8bf h[2]; };

// ---------------- elementwise helpers ----------------

__global__ __launch_bounds__(256) void f2bf_kernel(const float* __restrict__ in,
                                                   __bf16* __restrict__ out, int n) {
  int i = blockIdx.x * blockDim.x + threadIdx.x;
  if (i < n) out[i] = (__bf16)in[i];
}

__global__ __launch_bounds__(256) void zero_kernel(float* __restrict__ p, int n) {
  int i = blockIdx.x * blockDim.x + threadIdx.x;
  if (i < n) p[i] = 0.f;
}

// Slice dt_r = x_db[:, 0:64] and convert to bf16 (A-matrix of dt_proj GEMM).
__global__ __launch_bounds__(256) void take_dtr_kernel(const float* __restrict__ xdb,
                                                       __bf16* __restrict__ dtr) {
  int i = blockIdx.x * blockDim.x + threadIdx.x;
  if (i < NTOK * DTRANK) {
    int m = i >> 6, j = i & 63;
    dtr[i] = (__bf16)xdb[m * XDBW + j];
  }
}

// ---------------- generic WMMA GEMM ----------------
// C[m][n] = sum_k A[m][k] * W[n][k];  A:[M,K] bf16 row-major, W:[N,K] bf16
// row-major (i.e. computes A @ W^T, matching the reference's `@ w.T`).
// Each wave owns a 32x32 tile = 2x2 grid of 16x16x32 WMMA accumulators.
// MODE 0: plain f32 store.
// MODE 1: v = softplus(v + bias[n])        (dt_proj epilogue)
// MODE 2: store to C [token-major] and C2 [(l*B+b)-major]  (out_proj epilogue)
template <int MODE>
__global__ __launch_bounds__(256) void wmma_gemm_kernel(
    const __bf16* __restrict__ A, const __bf16* __restrict__ W,
    float* __restrict__ C, float* __restrict__ C2,
    const float* __restrict__ bias, int M, int N, int K)
{
  const int lane = threadIdx.x & 31;
  const int wave = threadIdx.x >> 5;
  const int tiles_n = N >> 5;
  const int tiles_m = M >> 5;
  int tile = blockIdx.x * 8 + wave;
  if (tile >= tiles_m * tiles_n) return;
  const int tm = (tile / tiles_n) << 5;
  const int tn = (tile % tiles_n) << 5;

  const int lmod = lane & 15;
  const int hi   = lane >> 4;   // lane group 0: lanes 0-15, group 1: 16-31

  // ISA 7.12.2 16-bit A 16x32: lanes<16 hold K {0..7,16..23}; lanes>=16 {8..15,24..31}
  const __bf16* Ar0 = A + (size_t)(tm + lmod) * K + (hi ? 8 : 0);
  const __bf16* Ar1 = Ar0 + (size_t)16 * K;
  // 16-bit B 32x16: lanes<16 hold K {0..7,16..23} of col lane; lanes>=16 hold K {8..15,24..31}
  const __bf16* Wc0 = W + (size_t)(tn + lmod) * K + (hi ? 8 : 0);
  const __bf16* Wc1 = Wc0 + (size_t)16 * K;

  v8f acc00 = {}, acc01 = {}, acc10 = {}, acc11 = {};

  for (int k0 = 0; k0 < K; k0 += 32) {
    AFrag a0, a1, b0, b1;
    a0.h[0] = *(const v8bf*)(Ar0 + k0);
    a0.h[1] = *(const v8bf*)(Ar0 + k0 + 16);
    a1.h[0] = *(const v8bf*)(Ar1 + k0);
    a1.h[1] = *(const v8bf*)(Ar1 + k0 + 16);
    b0.h[0] = *(const v8bf*)(Wc0 + k0);
    b0.h[1] = *(const v8bf*)(Wc0 + k0 + 16);
    b1.h[0] = *(const v8bf*)(Wc1 + k0);
    b1.h[1] = *(const v8bf*)(Wc1 + k0 + 16);
    acc00 = __builtin_amdgcn_wmma_f32_16x16x32_bf16(false, a0.v, false, b0.v, (short)0, acc00, false, false);
    acc01 = __builtin_amdgcn_wmma_f32_16x16x32_bf16(false, a0.v, false, b1.v, (short)0, acc01, false, false);
    acc10 = __builtin_amdgcn_wmma_f32_16x16x32_bf16(false, a1.v, false, b0.v, (short)0, acc10, false, false);
    acc11 = __builtin_amdgcn_wmma_f32_16x16x32_bf16(false, a1.v, false, b1.v, (short)0, acc11, false, false);
  }

  // C/D layout: VGPR r, lanes<16 -> (M=r, N=lane), lanes>=16 -> (M=r+8, N=lane-16)
  #pragma unroll
  for (int q = 0; q < 4; ++q) {
    const int mi = q >> 1, ni = q & 1;
    v8f a = (q == 0) ? acc00 : (q == 1) ? acc01 : (q == 2) ? acc10 : acc11;
    const int rb  = tm + mi * 16 + hi * 8;
    const int col = tn + ni * 16 + lmod;
    #pragma unroll
    for (int r = 0; r < 8; ++r) {
      float v = a[r];
      const int row = rb + r;
      if (MODE == 1) {
        v += bias[col];
        v = (v > 20.f) ? v : log1pf(__expf(v));   // softplus
      }
      C[(size_t)row * N + col] = v;
      if (MODE == 2) {
        const int bb = row >> 10;            // token = b*L + l
        const int ll = row & (SEQL - 1);
        C2[(size_t)((ll << 2) + bb) * N + col] = v;   // out_traj [L,B,D]
      }
    }
  }
}

// ---------------- depthwise conv + SiLU + conv_traj ----------------
__global__ __launch_bounds__(256) void conv_silu_kernel(
    const float* __restrict__ XZ, const float* __restrict__ conv_w,
    const float* __restrict__ conv_b,
    float* __restrict__ xc_f, __bf16* __restrict__ xc_bf,
    float* __restrict__ conv_traj)
{
  int idx = blockIdx.x * blockDim.x + threadIdx.x;
  if (idx >= NTOK * DINNER) return;
  const int d = idx % DINNER;
  const int m = idx / DINNER;
  const int b = m >> 10;
  const int l = m & (SEQL - 1);

  float xi[DCONV];
  #pragma unroll
  for (int k = 0; k < DCONV; ++k) {
    const int li = l - (DCONV - 1) + k;
    xi[k] = (li >= 0) ? XZ[(size_t)((b << 10) + li) * (2 * DINNER) + d] : 0.f;
  }
  float acc = conv_b[d];
  #pragma unroll
  for (int k = 0; k < DCONV; ++k) acc += xi[k] * conv_w[d * DCONV + k];
  const float s = acc / (1.f + __expf(-acc));      // silu
  xc_f[idx]  = s;
  xc_bf[idx] = (__bf16)s;

  float4* ct = (float4*)(conv_traj + (size_t)(l + 1) * (SEQB * DINNER * DCONV)
                                   + (size_t)b * (DINNER * DCONV) + (size_t)d * DCONV);
  *ct = make_float4(xi[0], xi[1], xi[2], xi[3]);
}

// ---------------- the only sequential part: SSM recurrence ----------------
// Thread per (b, d, s): 131072 threads = 4096 wave32s. Each lane carries one
// f32 state; per step the y-readout is reduced across the 16 state lanes of
// each half-wave with __shfl_xor. ssm_traj stores are 128B contiguous per
// wave. Next-step dt/xc are prefetched (lowers to global_prefetch_b8).
__global__ __launch_bounds__(256) void scan_kernel(
    const float* __restrict__ XZ, const float* __restrict__ xc_f,
    const float* __restrict__ xdb, const float* __restrict__ dt_f,
    const float* __restrict__ A_log, const float* __restrict__ D_param,
    float* __restrict__ ssm_traj, __bf16* __restrict__ y_bf)
{
  const int g = blockIdx.x * blockDim.x + threadIdx.x;
  if (g >= SEQB * DINNER * DSTATE) return;
  const int s = g & (DSTATE - 1);
  const int d = (g >> 4) & (DINNER - 1);
  const int b = g >> 15;

  const float An = -__expf(A_log[d * DSTATE + s]);
  const float Dp = D_param[d];
  float ss = 0.f;

  for (int l = 0; l < SEQL; ++l) {
    const int m = (b << 10) + l;
    const float dt = dt_f[(size_t)m * DINNER + d];
    const float xc = xc_f[(size_t)m * DINNER + d];
    const float z  = XZ [(size_t)m * (2 * DINNER) + DINNER + d];
    const float Bt = xdb[(size_t)m * XDBW + DTRANK + s];
    const float Ct = xdb[(size_t)m * XDBW + DTRANK + DSTATE + s];

    if (l + 1 < SEQL) {   // prefetch next step's recurrence operands
      __builtin_prefetch(dt_f + (size_t)(m + 1) * DINNER + d, 0, 1);
      __builtin_prefetch(xc_f + (size_t)(m + 1) * DINNER + d, 0, 1);
      __builtin_prefetch(xdb  + (size_t)(m + 1) * XDBW + DTRANK + s, 0, 1);
    }

    ss = ss * __expf(dt * An) + (dt * xc) * Bt;

    ssm_traj[(size_t)(l + 1) * (SEQB * DINNER * DSTATE)
             + (size_t)b * (DINNER * DSTATE) + d * DSTATE + s] = ss;

    float y = ss * Ct;
    y += __shfl_xor(y, 1, 32);   // reduce across the 16 state lanes
    y += __shfl_xor(y, 2, 32);
    y += __shfl_xor(y, 4, 32);
    y += __shfl_xor(y, 8, 32);
    if (s == 0) {
      float yy = y + Dp * xc;
      yy *= z / (1.f + __expf(-z));                    // * silu(z)
      y_bf[(size_t)m * DINNER + d] = (__bf16)yy;
    }
  }
}

// ---------------- launch ----------------
extern "C" void kernel_launch(void* const* d_in, const int* in_sizes, int n_in,
                              void* d_out, int out_size, void* d_ws, size_t ws_size,
                              hipStream_t stream)
{
  (void)in_sizes; (void)n_in; (void)out_size; (void)ws_size;
  const float* x          = (const float*)d_in[0];
  const float* in_proj_w  = (const float*)d_in[1];
  const float* conv_w     = (const float*)d_in[2];
  const float* conv_b     = (const float*)d_in[3];
  const float* x_proj_w   = (const float*)d_in[4];
  const float* dt_proj_w  = (const float*)d_in[5];
  const float* dt_proj_b  = (const float*)d_in[6];
  const float* A_log      = (const float*)d_in[7];
  const float* D_param    = (const float*)d_in[8];
  const float* out_proj_w = (const float*)d_in[9];

  float* out       = (float*)d_out;
  float* outputs   = out;                                              // [B,L,D]
  float* out_traj  = outputs  + (size_t)SEQB * SEQL * DMODEL;          // [L,B,D]
  float* conv_traj = out_traj + (size_t)SEQL * SEQB * DMODEL;          // [L+1,B,Din,4]
  float* ssm_traj  = conv_traj + (size_t)(SEQL + 1) * SEQB * DINNER * DCONV; // [L+1,B,Din,16]

  char* ws = (char*)d_ws;
  size_t off = 0;
  auto carve = [&](size_t bytes) -> void* {
    void* p = ws + off;
    off = (off + bytes + 255) & ~(size_t)255;
    return p;
  };
  __bf16* Xbf  = (__bf16*)carve((size_t)NTOK * DMODEL * 2);
  __bf16* Wi   = (__bf16*)carve((size_t)(2 * DINNER) * DMODEL * 2);
  float*  XZ   = (float*) carve((size_t)NTOK * (2 * DINNER) * 4);
  float*  xc_f = (float*) carve((size_t)NTOK * DINNER * 4);
  __bf16* xc_b = (__bf16*)carve((size_t)NTOK * DINNER * 2);
  __bf16* Wx   = (__bf16*)carve((size_t)XDBW * DINNER * 2);
  float*  xdb  = (float*) carve((size_t)NTOK * XDBW * 4);
  __bf16* dtr  = (__bf16*)carve((size_t)NTOK * DTRANK * 2);
  __bf16* Wdt  = (__bf16*)carve((size_t)DINNER * DTRANK * 2);
  float*  dt_f = (float*) carve((size_t)NTOK * DINNER * 4);
  __bf16* y_b  = (__bf16*)carve((size_t)NTOK * DINNER * 2);
  __bf16* Wo   = (__bf16*)carve((size_t)DMODEL * DINNER * 2);

  const int T = 256;
  auto nb = [](size_t n) { return (unsigned)((n + 255) / 256); };

  // bf16 conversions (weights + activations)
  f2bf_kernel<<<nb((size_t)NTOK * DMODEL), T, 0, stream>>>(x, Xbf, NTOK * DMODEL);
  f2bf_kernel<<<nb((size_t)(2 * DINNER) * DMODEL), T, 0, stream>>>(in_proj_w, Wi, 2 * DINNER * DMODEL);
  f2bf_kernel<<<nb((size_t)XDBW * DINNER), T, 0, stream>>>(x_proj_w, Wx, XDBW * DINNER);
  f2bf_kernel<<<nb((size_t)DINNER * DTRANK), T, 0, stream>>>(dt_proj_w, Wdt, DINNER * DTRANK);
  f2bf_kernel<<<nb((size_t)DMODEL * DINNER), T, 0, stream>>>(out_proj_w, Wo, DMODEL * DINNER);

  // trajectory t=0 slices are zeros
  zero_kernel<<<nb(SEQB * DINNER * DCONV), T, 0, stream>>>(conv_traj, SEQB * DINNER * DCONV);
  zero_kernel<<<nb(SEQB * DINNER * DSTATE), T, 0, stream>>>(ssm_traj, SEQB * DINNER * DSTATE);

  // GEMM1: XZ = X @ in_proj_w^T   [4096 x 4096, K=1024]
  { int tiles = (NTOK >> 5) * ((2 * DINNER) >> 5);
    wmma_gemm_kernel<0><<<(tiles + 7) / 8, T, 0, stream>>>(
        Xbf, Wi, XZ, nullptr, nullptr, NTOK, 2 * DINNER, DMODEL); }

  // depthwise conv + silu, writes conv_traj[1..L]
  conv_silu_kernel<<<nb((size_t)NTOK * DINNER), T, 0, stream>>>(
      XZ, conv_w, conv_b, xc_f, xc_b, conv_traj);

  // GEMM2: x_db = xc @ x_proj_w^T   [4096 x 96, K=2048]
  { int tiles = (NTOK >> 5) * (XDBW >> 5);
    wmma_gemm_kernel<0><<<(tiles + 7) / 8, T, 0, stream>>>(
        xc_b, Wx, xdb, nullptr, nullptr, NTOK, XDBW, DINNER); }

  take_dtr_kernel<<<nb((size_t)NTOK * DTRANK), T, 0, stream>>>(xdb, dtr);

  // GEMM3: dt = softplus(dt_r @ dt_proj_w^T + b)   [4096 x 2048, K=64]
  { int tiles = (NTOK >> 5) * (DINNER >> 5);
    wmma_gemm_kernel<1><<<(tiles + 7) / 8, T, 0, stream>>>(
        dtr, Wdt, dt_f, nullptr, dt_proj_b, NTOK, DINNER, DTRANK); }

  // sequential SSM recurrence, writes ssm_traj[1..L] and y (bf16)
  scan_kernel<<<nb((size_t)SEQB * DINNER * DSTATE), T, 0, stream>>>(
      XZ, xc_f, xdb, dt_f, A_log, D_param, ssm_traj, y_b);

  // GEMM4: out = y @ out_proj_w^T, dual-layout store (outputs + out_traj)
  { int tiles = (NTOK >> 5) * (DMODEL >> 5);
    wmma_gemm_kernel<2><<<(tiles + 7) / 8, T, 0, stream>>>(
        y_b, Wo, outputs, out_traj, nullptr, NTOK, DMODEL, DINNER); }
}
